// HMModel_21732534518130
// MI455X (gfx1250) — matrix operand order repaired and checked by reading, not compiled, not run on hardware
//
#include <hip/hip_runtime.h>
#include <hip/hip_bf16.h>

typedef __attribute__((ext_vector_type(16))) __bf16 bf16x16;
typedef __attribute__((ext_vector_type(8)))  float  f32x8;

#define V_N 100000
#define D_N 128
#define B_N 16
#define H_N 64
#define VT_N (V_N / 16)          // 6250 v-tiles
#define NWAVE (B_N * VT_N)       // 100000 waves
#define GBLOCKS (NWAVE / 8)      // 12500 blocks (8 waves each)
#define S_BLOCKS ((B_N * V_N) / 256) // 6250 blocks for pointwise passes

// ---------------- workspace layout (bytes) ----------------
static const size_t OFF_ASW = 0;                                   // 1024*128 bf16 = 256 KB
static const size_t OFF_BSW = 262144;                              // 100000*128 bf16 = 25.6 MB
static const size_t OFF_H1  = OFF_BSW + 25600000;                  // 16*8*100000 f32 = 51.2 MB
static const size_t OFF_P1  = OFF_H1 + 51200000;                   // 12500*16 f32
static const size_t OFF_P2  = OFF_P1 + 800000;                     // 6250*64 f32
static const size_t OFF_P3  = OFF_P2 + 1600000;                    // 6250*16 f32
static const size_t OFF_S1  = OFF_P3 + 400000;                     // 16 f32
static const size_t OFF_S2  = OFF_S1 + 256;                        // 64 f32
static const size_t OFF_S3  = OFF_S2 + 256;                        // 16 f32

__device__ __forceinline__ float leaky(float z) { return (z >= 0.f) ? z : 0.01f * z; }

// ---- P0: gather+normalize history rows, scatter into WMMA A-fragment layout ----
__global__ void p0_kernel(const int* __restrict__ hist, const float* __restrict__ emb,
                          __bf16* __restrict__ Asw) {
  __shared__ float red[128];
  const int r = blockIdx.x;          // 0..1023 = b*64+h
  const int b = r >> 6, h = r & 63;
  const int k = threadIdx.x;         // 0..127
  const int idx = hist[r];
  float val = emb[(size_t)idx * D_N + k];
  red[k] = val * val;
  __syncthreads();
  for (int s = 64; s > 0; s >>= 1) { if (k < s) red[k] += red[k + s]; __syncthreads(); }
  float a = val / fmaxf(sqrtf(red[0]), 1e-12f);
  // A 16x32 bf16 fragment mapping: lane = (h%16) + 16*half, K = (e<8?0:16) + 8*half + (e%8)
  int hc = h >> 4, M = h & 15, c = k >> 5, ko = k & 31;
  int g = (ko >> 4) & 1, hf = (ko >> 3) & 1, i = ko & 7;
  int e = 8 * g + i, lane = M + 16 * hf;
  Asw[(((size_t)(b * 4 + hc) * 4 + c) * 32 + lane) * 16 + e] = (__bf16)a;
}

// ---- P1: normalize all embedding rows into WMMA B-fragment layout (K x N tiles) ----
__global__ void p1_kernel(const float* __restrict__ emb, __bf16* __restrict__ Bsw) {
  __shared__ float red[128];
  const int v = blockIdx.x;          // 0..99999
  const int k = threadIdx.x;         // 0..127
  float val = emb[(size_t)v * D_N + k];
  red[k] = val * val;
  __syncthreads();
  for (int s = 64; s > 0; s >>= 1) { if (k < s) red[k] += red[k + s]; __syncthreads(); }
  float a = val / fmaxf(sqrtf(red[0]), 1e-12f);
  int c = k >> 5, ko = k & 31;
  int e = 8 * ((ko >> 4) & 1) + (ko & 7), ln = (v & 15) + 16 * ((ko >> 3) & 1);
  Bsw[(((size_t)(v >> 4) * 4 + c) * 32 + ln) * 16 + e] = (__bf16)a;
}

// ---- G: WMMA GEMM + fused max/argmax over h + logit/feat + conv1/leaky + BN1 partials ----
__global__ __launch_bounds__(256)
void g_kernel(const __bf16* __restrict__ Asw, const __bf16* __restrict__ Bsw,
              const float* __restrict__ week, const float* __restrict__ lik,
              const float* __restrict__ w1, const float* __restrict__ b1,
              float* __restrict__ h1, float* __restrict__ partials1) {
  __shared__ float s_part[16];
  const int wv = threadIdx.x >> 5;
  const int lane = threadIdx.x & 31;
  if (threadIdx.x < 16) s_part[threadIdx.x] = 0.f;
  __syncthreads();

  const int w = blockIdx.x * 8 + wv;
  const int b = w & 15;
  const int vt = w >> 4;

  // load all 16 A fragments (4 h-chunks x 4 k-chunks), fully coalesced
  bf16x16 afrag[16];
  const __bf16* abase = Asw + (size_t)b * 16 * 512 + (size_t)lane * 16;
#pragma unroll
  for (int f = 0; f < 16; ++f) afrag[f] = *(const bf16x16*)(abase + (size_t)f * 512);

  f32x8 acc[4] = {};
  const __bf16* bbase = Bsw + ((size_t)vt * 128 + lane) * 16;
#pragma unroll
  for (int c = 0; c < 4; ++c) {
    bf16x16 bfrag = *(const bf16x16*)(bbase + (size_t)c * 512);
#pragma unroll
    for (int hc = 0; hc < 4; ++hc)
      acc[hc] = __builtin_amdgcn_wmma_f32_16x16x32_bf16(
          false, afrag[hc * 4 + c], false, bfrag, (short)0, acc[hc], false, false);
  }

  // per-lane max over 32 of 64 h values (rows of D tiles), then swap halves
  const int half = lane >> 4;
  float best = -1e30f; int bestH = 0;
#pragma unroll
  for (int hc = 0; hc < 4; ++hc)
#pragma unroll
    for (int r = 0; r < 8; ++r) {
      float vv = acc[hc][r];
      int hh = hc * 16 + half * 8 + r;
      if (vv > best || (vv == best && hh < bestH)) { best = vv; bestH = hh; }
    }
  float ov = __shfl_xor(best, 16);
  int   oh = __shfl_xor(bestH, 16);
  if (ov > best || (ov == best && oh < bestH)) { best = ov; bestH = oh; }

  // epilogue: logit, week gather, likelihood, conv1 (3->8) + leaky
  float s = fminf(fmaxf(best, 0.001f), 0.999f);
  float logit = logf(s) - logf(1.f - s);
  float mw = week[b * H_N + bestH];
  const int v = vt * 16 + (lane & 15);
  float lk = lik[v];
  float hv[8];
#pragma unroll
  for (int ch = 0; ch < 8; ++ch) {
    float z = b1[ch] + w1[ch * 3] * logit + w1[ch * 3 + 1] * mw + w1[ch * 3 + 2] * lk;
    hv[ch] = leaky(z);
  }
  if (half == 0) {
#pragma unroll
    for (int ch = 0; ch < 8; ++ch)
      h1[(size_t)(b * 8 + ch) * V_N + v] = hv[ch];
  }
  // BN1 partial sums: reduce within each 16-lane half (halves hold duplicates)
#pragma unroll
  for (int ch = 0; ch < 8; ++ch) {
    float a = hv[ch], q = a * a;
#pragma unroll
    for (int m = 1; m <= 8; m <<= 1) { a += __shfl_xor(a, m); q += __shfl_xor(q, m); }
    if (lane == 0) { atomicAdd(&s_part[ch], a); atomicAdd(&s_part[8 + ch], q); }
  }
  __syncthreads();
  if (threadIdx.x < 16) partials1[(size_t)blockIdx.x * 16 + threadIdx.x] = s_part[threadIdx.x];
}

// ---- R: finish BN stats -> per-channel scale/shift ----
__global__ void r_kernel(const float* __restrict__ P, int nblk, int C,
                         const float* __restrict__ g, const float* __restrict__ be,
                         float* __restrict__ stats) {
  __shared__ float rs[256], rq[256];
  const int c = blockIdx.x, t = threadIdx.x;
  float s = 0.f, q = 0.f;
  for (int i = t; i < nblk; i += 256) {
    s += P[(size_t)i * 2 * C + c];
    q += P[(size_t)i * 2 * C + C + c];
  }
  rs[t] = s; rq[t] = q;
  __syncthreads();
  for (int st = 128; st > 0; st >>= 1) {
    if (t < st) { rs[t] += rs[t + st]; rq[t] += rq[t + st]; }
    __syncthreads();
  }
  if (t == 0) {
    const float N = (float)B_N * (float)V_N;
    float m = rs[0] / N;
    float var = rq[0] / N - m * m;
    float rsq = rsqrtf(var + 1e-5f);
    float sc = g[c] * rsq;
    stats[c] = sc;
    stats[C + c] = be[c] - m * sc;
  }
}

// ---- S2: bn1(h1) -> conv2(8->32)+leaky, accumulate BN2 partials (no h2 store) ----
__global__ __launch_bounds__(256)
void s2_kernel(const float* __restrict__ h1, const float* __restrict__ st1,
               const float* __restrict__ w2, const float* __restrict__ b2,
               float* __restrict__ partials2) {
  __shared__ float sw[256], sb[32], ss[16], s_part[64];
  const int t = threadIdx.x;
  sw[t] = w2[t];
  if (t < 32) sb[t] = b2[t];
  if (t < 16) ss[t] = st1[t];
  if (t < 64) s_part[t] = 0.f;
  __syncthreads();
  size_t n = (size_t)blockIdx.x * 256 + t;
  int b = (int)(n / V_N), v = (int)(n % V_N);
  float x[8];
#pragma unroll
  for (int c = 0; c < 8; ++c)
    x[c] = h1[(size_t)(b * 8 + c) * V_N + v] * ss[c] + ss[8 + c];
  const int lane = t & 31;
#pragma unroll
  for (int o = 0; o < 32; ++o) {
    float z = sb[o];
#pragma unroll
    for (int c = 0; c < 8; ++c) z += sw[o * 8 + c] * x[c];
    float a = leaky(z), s = a, q = a * a;
#pragma unroll
    for (int m = 1; m <= 16; m <<= 1) { s += __shfl_xor(s, m); q += __shfl_xor(q, m); }
    if (lane == 0) { atomicAdd(&s_part[o], s); atomicAdd(&s_part[32 + o], q); }
  }
  __syncthreads();
  if (t < 64) partials2[(size_t)blockIdx.x * 64 + t] = s_part[t];
}

// ---- S3: recompute h2 from h1, bn2 -> conv3(32->8)+leaky, BN3 partials ----
__global__ __launch_bounds__(256)
void s3_kernel(const float* __restrict__ h1, const float* __restrict__ st1,
               const float* __restrict__ w2, const float* __restrict__ b2,
               const float* __restrict__ st2,
               const float* __restrict__ w3, const float* __restrict__ b3,
               float* __restrict__ partials3) {
  __shared__ float sw2[256], sb2[32], ss1[16], ss2[64], sw3[256], sb3[8], s_part[16];
  const int t = threadIdx.x;
  sw2[t] = w2[t]; sw3[t] = w3[t];
  if (t < 32) sb2[t] = b2[t];
  if (t < 16) ss1[t] = st1[t];
  if (t < 64) ss2[t] = st2[t];
  if (t < 8)  sb3[t] = b3[t];
  if (t < 16) s_part[t] = 0.f;
  __syncthreads();
  size_t n = (size_t)blockIdx.x * 256 + t;
  int b = (int)(n / V_N), v = (int)(n % V_N);
  float x[8];
#pragma unroll
  for (int c = 0; c < 8; ++c)
    x[c] = h1[(size_t)(b * 8 + c) * V_N + v] * ss1[c] + ss1[8 + c];
  float y[8];
#pragma unroll
  for (int o3 = 0; o3 < 8; ++o3) y[o3] = sb3[o3];
#pragma unroll
  for (int o = 0; o < 32; ++o) {
    float z = sb2[o];
#pragma unroll
    for (int c = 0; c < 8; ++c) z += sw2[o * 8 + c] * x[c];
    float a = leaky(z) * ss2[o] + ss2[32 + o];
#pragma unroll
    for (int o3 = 0; o3 < 8; ++o3) y[o3] += sw3[o3 * 32 + o] * a;
  }
  const int lane = t & 31;
#pragma unroll
  for (int o3 = 0; o3 < 8; ++o3) {
    float a = leaky(y[o3]), s = a, q = a * a;
#pragma unroll
    for (int m = 1; m <= 16; m <<= 1) { s += __shfl_xor(s, m); q += __shfl_xor(q, m); }
    if (lane == 0) { atomicAdd(&s_part[o3], s); atomicAdd(&s_part[8 + o3], q); }
  }
  __syncthreads();
  if (t < 16) partials3[(size_t)blockIdx.x * 16 + t] = s_part[t];
}

// ---- F: recompute h2,h3 from h1, bn3 -> conv4(8->1) -> out ----
__global__ __launch_bounds__(256)
void f_kernel(const float* __restrict__ h1, const float* __restrict__ st1,
              const float* __restrict__ w2, const float* __restrict__ b2,
              const float* __restrict__ st2,
              const float* __restrict__ w3, const float* __restrict__ b3,
              const float* __restrict__ st3,
              const float* __restrict__ w4, const float* __restrict__ b4,
              float* __restrict__ out) {
  __shared__ float sw2[256], sb2[32], ss1[16], ss2[64], sw3[256], sb3[8], ss3[16], sw4[8], sb4;
  const int t = threadIdx.x;
  sw2[t] = w2[t]; sw3[t] = w3[t];
  if (t < 32) sb2[t] = b2[t];
  if (t < 16) ss1[t] = st1[t];
  if (t < 64) ss2[t] = st2[t];
  if (t < 8)  { sb3[t] = b3[t]; sw4[t] = w4[t]; }
  if (t < 16) ss3[t] = st3[t];
  if (t == 0) sb4 = b4[0];
  __syncthreads();
  size_t n = (size_t)blockIdx.x * 256 + t;
  int b = (int)(n / V_N), v = (int)(n % V_N);
  float x[8];
#pragma unroll
  for (int c = 0; c < 8; ++c)
    x[c] = h1[(size_t)(b * 8 + c) * V_N + v] * ss1[c] + ss1[8 + c];
  float y[8];
#pragma unroll
  for (int o3 = 0; o3 < 8; ++o3) y[o3] = sb3[o3];
#pragma unroll
  for (int o = 0; o < 32; ++o) {
    float z = sb2[o];
#pragma unroll
    for (int c = 0; c < 8; ++c) z += sw2[o * 8 + c] * x[c];
    float a = leaky(z) * ss2[o] + ss2[32 + o];
#pragma unroll
    for (int o3 = 0; o3 < 8; ++o3) y[o3] += sw3[o3 * 32 + o] * a;
  }
  float r = sb4;
#pragma unroll
  for (int o3 = 0; o3 < 8; ++o3)
    r += sw4[o3] * (leaky(y[o3]) * ss3[o3] + ss3[8 + o3]);
  out[n] = r;
}

extern "C" void kernel_launch(void* const* d_in, const int* in_sizes, int n_in,
                              void* d_out, int out_size, void* d_ws, size_t ws_size,
                              hipStream_t stream) {
  (void)in_sizes; (void)n_in; (void)out_size; (void)ws_size;
  const int*   hist = (const int*)  d_in[0];
  const float* week = (const float*)d_in[1];
  const float* emb  = (const float*)d_in[2];
  const float* lik  = (const float*)d_in[3];
  const float* w1 = (const float*)d_in[4];  const float* b1 = (const float*)d_in[5];
  const float* g1 = (const float*)d_in[6];  const float* be1= (const float*)d_in[7];
  const float* w2 = (const float*)d_in[8];  const float* b2 = (const float*)d_in[9];
  const float* g2 = (const float*)d_in[10]; const float* be2= (const float*)d_in[11];
  const float* w3 = (const float*)d_in[12]; const float* b3 = (const float*)d_in[13];
  const float* g3 = (const float*)d_in[14]; const float* be3= (const float*)d_in[15];
  const float* w4 = (const float*)d_in[16]; const float* b4 = (const float*)d_in[17];

  char* ws = (char*)d_ws;
  __bf16* Asw = (__bf16*)(ws + OFF_ASW);
  __bf16* Bsw = (__bf16*)(ws + OFF_BSW);
  float* h1  = (float*)(ws + OFF_H1);
  float* p1b = (float*)(ws + OFF_P1);
  float* p2b = (float*)(ws + OFF_P2);
  float* p3b = (float*)(ws + OFF_P3);
  float* st1 = (float*)(ws + OFF_S1);
  float* st2 = (float*)(ws + OFF_S2);
  float* st3 = (float*)(ws + OFF_S3);

  p0_kernel<<<B_N * H_N, 128, 0, stream>>>(hist, emb, Asw);
  p1_kernel<<<V_N, 128, 0, stream>>>(emb, Bsw);
  g_kernel<<<GBLOCKS, 256, 0, stream>>>(Asw, Bsw, week, lik, w1, b1, h1, p1b);
  r_kernel<<<8, 256, 0, stream>>>(p1b, GBLOCKS, 8, g1, be1, st1);
  s2_kernel<<<S_BLOCKS, 256, 0, stream>>>(h1, st1, w2, b2, p2b);
  r_kernel<<<32, 256, 0, stream>>>(p2b, S_BLOCKS, 32, g2, be2, st2);
  s3_kernel<<<S_BLOCKS, 256, 0, stream>>>(h1, st1, w2, b2, st2, w3, b3, p3b);
  r_kernel<<<8, 256, 0, stream>>>(p3b, S_BLOCKS, 8, g3, be3, st3);
  f_kernel<<<S_BLOCKS, 256, 0, stream>>>(h1, st1, w2, b2, st2, w3, b3, st3, w4, b4,
                                         (float*)d_out);
}
